// Model_2_50826642981713
// MI455X (gfx1250) — matrix operand order repaired and checked by reading, not compiled, not run on hardware
//
#include <hip/hip_runtime.h>
#include <cstdint>
#include <cmath>

// ---------------- WMMA types (CDNA5 gfx1250, wave32) ----------------
typedef __attribute__((ext_vector_type(16))) __bf16 v16bf;
typedef __attribute__((ext_vector_type(8)))  float  v8f;

union FragAB {
    v16bf    v;
    __bf16   h[16];
    uint32_t u[8];
};

#define S_LEN 2048
#define DINP  512          // 496 padded to 512
#define DIN_R 496
#define H4    3072
#define HID   768
#define H2    1536
#define NC    13
#define NCP   16
#define NWG   24           // workgroups per direction in the LSTM scan (8 waves each -> 192 tiles)

__device__ __forceinline__ float sigm(float x) { return 1.0f / (1.0f + expf(-x)); }

// ---------------- weight conversion kernels ----------------
__global__ void conv_wih(const float* __restrict__ Wf, const float* __restrict__ Wb,
                         __bf16* __restrict__ out) {
    size_t i = (size_t)blockIdx.x * blockDim.x + threadIdx.x;   // [2][3072][512]
    if (i >= (size_t)2 * H4 * DINP) return;
    int k = (int)(i & (DINP - 1));
    size_t nd = i >> 9;                // dir*3072 + n
    int n = (int)(nd % H4);
    int dir = (int)(nd / H4);
    const float* W = dir ? Wb : Wf;
    out[i] = (k < DIN_R) ? (__bf16)W[(size_t)n * DIN_R + k] : (__bf16)0.0f;
}

__global__ void conv_whh(const float* __restrict__ Wf, const float* __restrict__ Wb,
                         __bf16* __restrict__ out) {
    size_t i = (size_t)blockIdx.x * blockDim.x + threadIdx.x;   // [2][3072][768]
    if (i >= (size_t)2 * H4 * HID) return;
    size_t per = (size_t)H4 * HID;
    int dir = (int)(i / per);
    size_t rem = i - (size_t)dir * per;
    out[i] = (__bf16)((dir ? Wb : Wf)[rem]);
}

__global__ void conv_wc(const float* __restrict__ Wc, __bf16* __restrict__ out) {
    int i = blockIdx.x * blockDim.x + threadIdx.x;              // [16][1536]
    if (i >= NCP * H2) return;
    int n = i / H2, k = i - n * H2;
    out[i] = (n < NC) ? (__bf16)Wc[(size_t)n * H2 + k] : (__bf16)0.0f;
}

__global__ void zero_ints(int* __restrict__ p, int n) {
    int i = blockIdx.x * blockDim.x + threadIdx.x;
    if (i < n) p[i] = 0;
}

// ---------------- embedding gather + bf16 convert ----------------
__global__ void embed_gather(const int* __restrict__ x, const int* __restrict__ casing,
                             const int* __restrict__ pos,
                             const float* __restrict__ word_emb,
                             const float* __restrict__ casing_emb,
                             const float* __restrict__ pos_emb,
                             __bf16* __restrict__ embbf) {      // [S][512]
    int s = blockIdx.x;
    int k = threadIdx.x;
    float v;
    if (k < 400)      v = word_emb[(size_t)x[s] * 400 + k];
    else if (k < 432) v = casing_emb[(size_t)casing[s] * 32 + (k - 400)];
    else if (k < 496) v = pos_emb[(size_t)pos[s] * 64 + (k - 432)];
    else              v = 0.0f;
    embbf[(size_t)s * DINP + k] = (__bf16)v;
}

// ---------------- input-to-hidden GEMM: xg = emb @ Wih^T + b ----------------
// A = emb (S x 512 bf16), B(k,n) = Wih[n][k] (n-major bf16), C accum f32.
__global__ void __launch_bounds__(256)
gemm_ih(const __bf16* __restrict__ embbf,   // [S][512]
        const __bf16* __restrict__ Wihbf,   // [2][3072][512]
        const float* __restrict__ b_f, const float* __restrict__ b_b,
        float* __restrict__ xg) {           // [2][S][3072]
    const int lane = threadIdx.x & 31;
    const int wave = threadIdx.x >> 5;
    const int n0 = (blockIdx.x * 8 + wave) * 16;
    const int m0 = blockIdx.y * 16;
    const int dir = blockIdx.z;

    const __bf16* W = Wihbf + (size_t)dir * H4 * DINP;
    const float* bias = dir ? b_b : b_f;
    float* out = xg + (size_t)dir * S_LEN * H4;

    const int mA = m0 + (lane & 15);
    const int nB = n0 + (lane & 15);
    const int hA = (lane >> 4) << 3;    // A: half-lane K offset (8)
    const int hB = (lane >> 4) << 4;    // unused; B loads also use hA's offset (8)

    v8f acc = {};
    for (int kb = 0; kb < DINP; kb += 32) {
        FragAB A, B;
#pragma unroll
        for (int v = 0; v < 8; ++v) {
            int ka = kb + ((v >> 2) << 4) + hA + ((v & 3) << 1);
            A.u[v] = *(const uint32_t*)(embbf + (size_t)mA * DINP + ka);
            int kk = kb + ((v >> 2) << 4) + hA + ((v & 3) << 1);
            B.u[v] = *(const uint32_t*)(W + (size_t)nB * DINP + kk);
        }
        acc = __builtin_amdgcn_wmma_f32_16x16x32_bf16(false, A.v, false, B.v,
                                                      (short)0, acc, false, false);
    }
    const int col = n0 + (lane & 15);
    const float bb = bias[col];
#pragma unroll
    for (int v = 0; v < 8; ++v) {
        int row = m0 + ((lane >> 4) << 3) + v;
        out[(size_t)row * H4 + col] = acc[v] + bb;
    }
}

// ---------------- multi-WGP sequential bidirectional LSTM scan ----------------
// NWG workgroups (8 waves each) per direction; each wave owns one 16-wide N-tile
// of y = Whh @ h. Steps synchronize with a device-scope atomic barrier (one per
// step). Each WG redundantly recomputes the gate math (deterministic f32), so h/c
// state stays local in LDS and a single barrier per step suffices. y is double-
// buffered on step parity so no WG can overwrite a buffer still being read.
__global__ void __launch_bounds__(256)
lstm_scan_mw(const float* __restrict__ xg,      // [2][S][3072]
             const __bf16* __restrict__ Whhbf,  // [2][3072][768]
             const float* __restrict__ h0, const float* __restrict__ c0,
             float* __restrict__ hs,            // [2][S][768]
             float* __restrict__ ybuf,          // [2][2][3072]
             int* __restrict__ barcnt) {        // [2][S]
    __shared__ __bf16 h_bf[HID];
    __shared__ float  c_f[HID];

    const int dir = blockIdx.x / NWG;
    const int wg  = blockIdx.x % NWG;
    const int tid = threadIdx.x;
    const int lane = tid & 31;
    const int wave = tid >> 5;

    const float*  xgd = xg + (size_t)dir * S_LEN * H4;
    const __bf16* W   = Whhbf + (size_t)dir * H4 * HID;
    float*        hsd = hs + (size_t)dir * S_LEN * HID;
    int*          bar = barcnt + dir * S_LEN;

#pragma unroll
    for (int r = 0; r < 3; ++r) {
        int j = tid + (r << 8);
        c_f[j]  = c0[dir * HID + j];
        h_bf[j] = (__bf16)h0[dir * HID + j];
    }
    __syncthreads();

    const int n0 = (wg * 8 + wave) * 16;          // this wave's N-tile of y
    const int nB = n0 + (lane & 15);
    const int hA = (lane >> 4) << 3;
    const int hB = (lane >> 4) << 4;

    for (int t = 0; t < S_LEN; ++t) {
        const int s = dir ? (S_LEN - 1 - t) : t;
        float* y = ybuf + ((size_t)(dir * 2 + (t & 1))) * H4;

        // --- y tile: 24 chained WMMAs over K=768, h broadcast from LDS ---
        v8f acc = {};
#pragma unroll 1
        for (int kb = 0; kb < HID; kb += 32) {
            FragAB A, B;
#pragma unroll
            for (int v = 0; v < 8; ++v) {
                int ka = kb + ((v >> 2) << 4) + hA + ((v & 3) << 1);
                A.u[v] = *(const uint32_t*)&h_bf[ka];          // LDS broadcast
                int kk = kb + ((v >> 2) << 4) + hA + ((v & 3) << 1);
                B.u[v] = *(const uint32_t*)(W + (size_t)nB * HID + kk);
            }
            acc = __builtin_amdgcn_wmma_f32_16x16x32_bf16(false, A.v, false, B.v,
                                                          (short)0, acc, false, false);
        }
        if (lane < 16) y[n0 + lane] = acc[0];  // all D rows equal (A rows replicated)

        // --- device-scope barrier for this step ---
        __threadfence();   // release: make our y stores visible device-wide
        __syncthreads();
        if (tid == 0) {
            __hip_atomic_fetch_add(&bar[t], 1, __ATOMIC_RELAXED, __HIP_MEMORY_SCOPE_AGENT);
            while (__hip_atomic_load(&bar[t], __ATOMIC_RELAXED, __HIP_MEMORY_SCOPE_AGENT) < NWG)
                __builtin_amdgcn_s_sleep(1);
        }
        __syncthreads();
        __threadfence();   // acquire: observe other WGs' y stores

        // --- gate math, recomputed redundantly in every WG (768 lanes) ---
        const float* xr = xgd + (size_t)s * H4;
#pragma unroll
        for (int r = 0; r < 3; ++r) {
            int j = tid + (r << 8);
            float ig = xr[j]           + y[j];
            float fg = xr[HID + j]     + y[HID + j];
            float gg = xr[2 * HID + j] + y[2 * HID + j];
            float og = xr[3 * HID + j] + y[3 * HID + j];
            float c  = sigm(fg) * c_f[j] + sigm(ig) * tanhf(gg);
            float h  = sigm(og) * tanhf(c);
            c_f[j]  = c;
            h_bf[j] = (__bf16)h;
            if (wg == 0) hsd[(size_t)s * HID + j] = h;
        }
        __syncthreads();
    }
}

// ---------------- output projection: o = [hs_f, hs_b] @ Wc^T + bc ----------------
__global__ void __launch_bounds__(32)
gemm_out(const float* __restrict__ hs,      // [2][S][768]
         const __bf16* __restrict__ Wcbf,   // [16][1536] (rows >=13 zero)
         const float* __restrict__ bc,
         float* __restrict__ o) {           // [S][16]
    const int lane = threadIdx.x & 31;
    const int m0 = blockIdx.x * 16;
    const int mA = m0 + (lane & 15);
    const int nB = lane & 15;
    const int hA = (lane >> 4) << 3;
    const int hB = (lane >> 4) << 4;
    const float* hsf = hs;
    const float* hsb = hs + (size_t)S_LEN * HID;

    v8f acc = {};
    for (int kb = 0; kb < H2; kb += 32) {
        FragAB A, B;
#pragma unroll
        for (int v = 0; v < 8; ++v) {
            int ka = kb + ((v >> 2) << 4) + hA + ((v & 3) << 1);
            const float* src = (ka < HID) ? (hsf + (size_t)mA * HID + ka)
                                          : (hsb + (size_t)mA * HID + (ka - HID));
            A.h[2 * v]     = (__bf16)src[0];
            A.h[2 * v + 1] = (__bf16)src[1];
            int kk = kb + ((v >> 2) << 4) + hA + ((v & 3) << 1);
            B.u[v] = *(const uint32_t*)(Wcbf + (size_t)nB * H2 + kk);
        }
        acc = __builtin_amdgcn_wmma_f32_16x16x32_bf16(false, A.v, false, B.v,
                                                      (short)0, acc, false, false);
    }
    const float bb = (nB < NC) ? bc[nB] : 0.0f;
#pragma unroll
    for (int v = 0; v < 8; ++v) {
        int row = m0 + ((lane >> 4) << 3) + v;
        o[(size_t)row * NCP + nB] = acc[v] + bb;
    }
}

// ---------------- log_softmax over 13 classes ----------------
__global__ void log_softmax_rows(const float* __restrict__ o,   // [S][16]
                                 float* __restrict__ ts) {      // [S][13]
    int row = blockIdx.x * blockDim.x + threadIdx.x;
    if (row >= S_LEN) return;
    const float* r = o + (size_t)row * NCP;
    float m = -1e30f;
#pragma unroll
    for (int c = 0; c < NC; ++c) m = fmaxf(m, r[c]);
    float sum = 0.0f;
#pragma unroll
    for (int c = 0; c < NC; ++c) sum += expf(r[c] - m);
    float l = logf(sum);
#pragma unroll
    for (int c = 0; c < NC; ++c) ts[(size_t)row * NC + c] = r[c] - m - l;
}

// ---------------- CRF Viterbi decode ----------------
__global__ void __launch_bounds__(256)
viterbi(const float* __restrict__ ts,          // [S][13] tag_scores
        const float* __restrict__ start_trans, const float* __restrict__ end_trans,
        const float* __restrict__ trans,       // [13][13]
        int* __restrict__ bp,                  // [S][13]
        float* __restrict__ tags) {            // [S]
    __shared__ float tr[NC * NC];
    __shared__ float sc[NC];
    int tid = threadIdx.x;
    if (tid < NC * NC) tr[tid] = trans[tid];
    if (tid < NC) sc[tid] = start_trans[tid] + ts[tid];
    __syncthreads();

    for (int t = 1; t < S_LEN; ++t) {
        float best = -1e30f;
        int bpi = 0;
        if (tid < NC) {
            for (int p = 0; p < NC; ++p) {
                float v = sc[p] + tr[p * NC + tid];
                if (v > best) { best = v; bpi = p; }   // strict >: first-max like argmax
            }
        }
        __syncthreads();
        if (tid < NC) {
            sc[tid] = best + ts[(size_t)t * NC + tid];
            bp[(size_t)t * NC + tid] = bpi;
        }
        __syncthreads();
    }
    if (tid == 0) {
        float best = -1e30f;
        int last = 0;
        for (int p = 0; p < NC; ++p) {
            float v = sc[p] + end_trans[p];
            if (v > best) { best = v; last = p; }
        }
        tags[S_LEN - 1] = (float)last;
        for (int t = S_LEN - 1; t >= 1; --t) {
            last = bp[(size_t)t * NC + last];
            tags[t - 1] = (float)last;
        }
    }
}

// ---------------- host launcher ----------------
extern "C" void kernel_launch(void* const* d_in, const int* in_sizes, int n_in,
                              void* d_out, int out_size, void* d_ws, size_t ws_size,
                              hipStream_t stream) {
    const int*   x           = (const int*)d_in[0];
    const int*   casing      = (const int*)d_in[1];
    const int*   pos         = (const int*)d_in[2];
    const float* word_emb    = (const float*)d_in[3];
    const float* casing_emb  = (const float*)d_in[4];
    const float* pos_emb     = (const float*)d_in[5];
    const float* Wih_f       = (const float*)d_in[6];
    const float* Whh_f       = (const float*)d_in[7];
    const float* b_f         = (const float*)d_in[8];
    const float* Wih_b       = (const float*)d_in[9];
    const float* Whh_b       = (const float*)d_in[10];
    const float* b_b         = (const float*)d_in[11];
    const float* Wc          = (const float*)d_in[12];
    const float* bc          = (const float*)d_in[13];
    const float* start_trans = (const float*)d_in[14];
    const float* end_trans   = (const float*)d_in[15];
    const float* trans       = (const float*)d_in[16];
    const float* h0          = (const float*)d_in[17];
    const float* c0          = (const float*)d_in[18];

    char* p = (char*)d_ws;
    auto alloc = [&](size_t bytes) -> void* {
        void* r = (void*)p;
        p += (bytes + 255) & ~(size_t)255;
        return r;
    };
    __bf16* embbf  = (__bf16*)alloc((size_t)S_LEN * DINP * 2);
    __bf16* Wihbf  = (__bf16*)alloc((size_t)2 * H4 * DINP * 2);
    __bf16* Whhbf  = (__bf16*)alloc((size_t)2 * H4 * HID * 2);
    __bf16* Wcbf   = (__bf16*)alloc((size_t)NCP * H2 * 2);
    float*  xgws   = (float*)alloc((size_t)2 * S_LEN * H4 * 4);
    float*  hsws   = (float*)alloc((size_t)2 * S_LEN * HID * 4);
    float*  ows    = (float*)alloc((size_t)S_LEN * NCP * 4);
    int*    bpws   = (int*)alloc((size_t)S_LEN * NC * 4);
    float*  ybuf   = (float*)alloc((size_t)2 * 2 * H4 * 4);
    int*    barcnt = (int*)alloc((size_t)2 * S_LEN * 4);

    float* tag_scores = (float*)d_out;
    float* tags = tag_scores + (size_t)S_LEN * NC;

    // weight conversion + barrier-counter init
    {
        size_t n = (size_t)2 * H4 * DINP;
        conv_wih<<<dim3((unsigned)((n + 255) / 256)), dim3(256), 0, stream>>>(Wih_f, Wih_b, Wihbf);
    }
    {
        size_t n = (size_t)2 * H4 * HID;
        conv_whh<<<dim3((unsigned)((n + 255) / 256)), dim3(256), 0, stream>>>(Whh_f, Whh_b, Whhbf);
    }
    conv_wc<<<dim3((NCP * H2 + 255) / 256), dim3(256), 0, stream>>>(Wc, Wcbf);
    zero_ints<<<dim3((2 * S_LEN + 255) / 256), dim3(256), 0, stream>>>(barcnt, 2 * S_LEN);

    // embedding gather
    embed_gather<<<dim3(S_LEN), dim3(DINP), 0, stream>>>(x, casing, pos, word_emb,
                                                         casing_emb, pos_emb, embbf);

    // input-to-hidden gates for both directions
    gemm_ih<<<dim3(H4 / (16 * 8), S_LEN / 16, 2), dim3(256), 0, stream>>>(
        embbf, Wihbf, b_f, b_b, xgws);

    // sequential bidirectional LSTM spread over 2*NWG workgroups
    lstm_scan_mw<<<dim3(2 * NWG), dim3(256), 0, stream>>>(xgws, Whhbf, h0, c0,
                                                          hsws, ybuf, barcnt);

    // projection + log_softmax
    gemm_out<<<dim3(S_LEN / 16), dim3(32), 0, stream>>>(hsws, Wcbf, bc, ows);
    log_softmax_rows<<<dim3(S_LEN / 256), dim3(256), 0, stream>>>(ows, tag_scores);

    // CRF Viterbi
    viterbi<<<dim3(1), dim3(256), 0, stream>>>(tag_scores, start_trans, end_trans,
                                               trans, bpws, tags);
}
